// ImplicitPointHead_27728308862990
// MI455X (gfx1250) — matrix operand order repaired and checked by reading, not compiled, not run on hardware
//
#include <hip/hip_runtime.h>
#include <hip/hip_bf16.h>

// ---------------------------------------------------------------------------
// ImplicitPointHead on MI455X (gfx1250): per-instance MLP via bf16 WMMA.
//   X[512x2048] -> L0[256x512] -> ReLU -> L1[256x256] -> ReLU
//             -> L2[256x256] -> ReLU -> L3[1x256]  (f32 accumulate)
// Grid: (16 point-tiles of 128, 128 instances), 256 threads = 8 wave32.
// Software-pipelined: A fragments double-buffered across K-steps (global/L2),
// B fragments 4-deep across N-tiles. B addressing uses two LDS base registers
// so every ds_load_b128 is [base + imm16] -> no inner-loop VALU address math
// -> no WMMA->VALU WAR v_nops.
// ---------------------------------------------------------------------------

typedef __attribute__((ext_vector_type(16))) __bf16 v16bf;
typedef __attribute__((ext_vector_type(8)))  __bf16 v8bf;
typedef __attribute__((ext_vector_type(8)))  float  v8f;

union ABFrag { v16bf v; v8bf h[2]; };

#define NUM_PARAMS 263169
#define W0_OFF 0
#define W1_OFF 131072
#define W2_OFF 196608
#define W3_OFF 262144
#define B0_OFF 262400
#define B1_OFF 262656
#define B2_OFF 262912
#define B3_OFF 263168

#define PTILE 128           // points per workgroup
#define LDS_X_HALFS (128*512)
#define LDS_Y_HALFS (128*256)
#define LDS_BYTES ((LDS_X_HALFS + LDS_Y_HALFS) * 2)

__device__ __forceinline__ v8bf cvt8_f32_bf16(const float* __restrict__ p) {
    v8bf r;
#pragma unroll
    for (int i = 0; i < 8; ++i) r[i] = (__bf16)p[i];
    return r;
}

// One dense layer: Y[256 x 128pts] = relu(W[256 x din] * X[din x 128pts] + b)
// X staged in LDS transposed: Xs[pt*din + ch] (bf16). Output written the same
// way at stride 256. Each wave owns 2 M-tiles x 8 N-tiles of 16x16.
__device__ __forceinline__ void mlp_layer(const float* __restrict__ Wg,
                                          const float* __restrict__ Bg,
                                          const __bf16* __restrict__ Xs, int din,
                                          __bf16* __restrict__ Ys, bool do_relu)
{
    const int tid  = threadIdx.x;
    const int lane = tid & 31;
    const int wave = tid >> 5;          // 0..7
    const int lr   = lane & 15;         // row (A) / col (B,C) within tile
    const int hi   = (lane >> 4) & 1;   // lane-group select

    v8f acc[2][8];
#pragma unroll
    for (int i = 0; i < 2; ++i)
#pragma unroll
        for (int n = 0; n < 8; ++n) { v8f z = {}; acc[i][n] = z; }

    // Per-lane base pointers.
    // A (16-bit 16x32): lanes 0-15 hold K 0-15, lanes 16-31 K 16-31.
    const float* wrow[2];
    wrow[0] = Wg + ((wave * 2 + 0) * 16 + lr) * din + hi * 16;
    wrow[1] = Wg + ((wave * 2 + 1) * 16 + lr) * din + hi * 16;
    // B (16-bit 32x16): lanes 0-15: K 0-15, lanes 16-31: K 16-31, col = lane%16.
    // Two LDS bases so all 8 N-tile offsets stay under the 16-bit DS immediate
    // (L0 stride 512 halves -> 16 KB per N-tile, tiles 4..7 would need VALU
    // address adds otherwise, which trip the WMMA->VALU WAR hazard).
    const __bf16* xb[2];
    xb[0] = Xs + lr * din + hi * 16;
    xb[1] = xb[0] + 4 * 16 * din;

    const int ksteps = din >> 5;        // K chunks of 32 (bf16 WMMA depth)

    // Prologue: A fragments for ks = 0.
    ABFrag aC[2], aN[2];
#pragma unroll
    for (int i = 0; i < 2; ++i) {
        aC[i].h[0] = cvt8_f32_bf16(wrow[i]);
        aC[i].h[1] = cvt8_f32_bf16(wrow[i] + 8);
    }

    for (int ks = 0; ks < ksteps; ++ks) {
        const int kofs = ks * 32;

        // Preload next K-step's A fragments (global/L2, long latency) so they
        // fly across the whole 16-WMMA inner loop below.
        if (ks + 1 < ksteps) {
#pragma unroll
            for (int i = 0; i < 2; ++i) {
                const float* wp = wrow[i] + (ks + 1) * 32;
                aN[i].h[0] = cvt8_f32_bf16(wp);
                aN[i].h[1] = cvt8_f32_bf16(wp + 8);
                __builtin_prefetch(wp + 32, 0, 3);   // ks+2 -> global_prefetch_b8
            }
        }

        // B pipeline: 4 slots, 2 N-tiles in flight. All loads are
        // ds_load_b128 [vBase + imm] (k-loop fully unrolled -> kofs folds).
        ABFrag b[4];
        {
            const v8bf* bp0 = (const v8bf*)(xb[0] + kofs);
            b[0].h[0] = bp0[0];
            b[0].h[1] = bp0[1];
            const v8bf* bp1 = (const v8bf*)(xb[0] + 16 * din + kofs);
            b[1].h[0] = bp1[0];
            b[1].h[1] = bp1[1];
        }
#pragma unroll
        for (int n = 0; n < 8; ++n) {
            if (n + 2 < 8) {
                const int m = n + 2;
                const v8bf* bp =
                    (const v8bf*)(xb[m >> 2] + (m & 3) * 16 * din + kofs);
                b[m & 3].h[0] = bp[0];
                b[m & 3].h[1] = bp[1];
            }
            acc[0][n] = __builtin_amdgcn_wmma_f32_16x16x32_bf16(
                false, aC[0].v, false, b[n & 3].v, (short)0, acc[0][n], false, false);
            acc[1][n] = __builtin_amdgcn_wmma_f32_16x16x32_bf16(
                false, aC[1].v, false, b[n & 3].v, (short)0, acc[1][n], false, false);
        }
#pragma unroll
        for (int i = 0; i < 2; ++i) aC[i] = aN[i];
    }

    // ---- epilogue: bias + ReLU, pack bf16, store transposed ----
    // C/D 16x16 f32 layout: VGPR r, lane l -> M = r + 8*(l/16), N = l%16.
#pragma unroll
    for (int i = 0; i < 2; ++i) {
        const int mbase = (wave * 2 + i) * 16 + hi * 8;
        float bias[8];
#pragma unroll
        for (int r = 0; r < 8; ++r) bias[r] = Bg[mbase + r];
#pragma unroll
        for (int n = 0; n < 8; ++n) {
            const int pt = n * 16 + lr;
            v8bf o;
#pragma unroll
            for (int r = 0; r < 8; ++r) {
                float v = acc[i][n][r] + bias[r];
                if (do_relu) v = v > 0.f ? v : 0.f;
                o[r] = (__bf16)v;
            }
            *(v8bf*)(Ys + pt * 256 + mbase) = o;   // ds_store_b128 [base + imm]
        }
    }
}

__global__ void __launch_bounds__(256, 1)
ImplicitPointHead_kernel(const float* __restrict__ feat,     // [128,256,2048]
                         const float* __restrict__ coords,   // [128,2048,2]
                         const float* __restrict__ params,   // [128,263169]
                         const float* __restrict__ gauss,    // [2,128]
                         float* __restrict__ out)            // [128,1,2048]
{
    extern __shared__ char smem[];
    __bf16* Xs = (__bf16*)smem;                      // [128][512] (L0), [128][256] later
    __bf16* Ys = (__bf16*)(smem + LDS_X_HALFS * 2);  // [128][256]

    const int tid  = threadIdx.x;
    const int inst = blockIdx.y;
    const int pt0  = blockIdx.x * PTILE;

    const float* P = params + (size_t)inst * NUM_PARAMS;

    // ---- stage layer-0 input transposed into LDS: Xs[pt*512 + ch] ----
    // channels 0..127 : sin PE, 128..255 : cos PE, 256..511 : image features
    const float TWO_PI = 6.283185307179586f;
    for (int idx = tid; idx < PTILE * 128; idx += 256) {
        const int pt = idx >> 7;
        const int c  = idx & 127;
        const float x = coords[((size_t)inst * 2048 + pt0 + pt) * 2 + 0];
        const float y = coords[((size_t)inst * 2048 + pt0 + pt) * 2 + 1];
        const float loc = (2.f * x - 1.f) * gauss[c] + (2.f * y - 1.f) * gauss[128 + c];
        const float a = TWO_PI * loc;
        Xs[pt * 512 + c]       = (__bf16)__sinf(a);
        Xs[pt * 512 + 128 + c] = (__bf16)__cosf(a);
    }
    const float* F = feat + (size_t)inst * 256 * 2048;
    for (int idx = tid; idx < 256 * PTILE; idx += 256) {
        const int c  = idx >> 7;
        const int pt = idx & 127;                    // consecutive tid -> coalesced pts
        Xs[pt * 512 + 256 + c] = (__bf16)F[c * 2048 + pt0 + pt];
    }
    __syncthreads();

    // ---- MLP trunk: WMMA layers (ping-pong LDS, both at stride 256) ----
    mlp_layer(P + W0_OFF, P + B0_OFF, Xs, 512, Ys, true);
    __syncthreads();
    mlp_layer(P + W1_OFF, P + B1_OFF, Ys, 256, Xs, true);  // Xs reused as [128][256]
    __syncthreads();
    mlp_layer(P + W2_OFF, P + B2_OFF, Xs, 256, Ys, true);
    __syncthreads();

    // ---- L3: out[pt] = dot(W3, Y2[:,pt]) + b3  (VALU; 1 output channel) ----
    if (tid < PTILE) {
        const float* W3 = P + W3_OFF;                // uniform -> scalar loads
        float acc3 = P[B3_OFF];
        const v8bf* yrow = (const v8bf*)(Ys + tid * 256);
#pragma unroll 4
        for (int c8 = 0; c8 < 32; ++c8) {
            const v8bf y = yrow[c8];                 // ds_load_b128
#pragma unroll
            for (int r = 0; r < 8; ++r) acc3 += (float)y[r] * W3[c8 * 8 + r];
        }
        out[(size_t)inst * 2048 + pt0 + tid] = acc3;
    }
}

extern "C" void kernel_launch(void* const* d_in, const int* in_sizes, int n_in,
                              void* d_out, int out_size, void* d_ws, size_t ws_size,
                              hipStream_t stream) {
    (void)in_sizes; (void)n_in; (void)out_size; (void)d_ws; (void)ws_size;
    const float* feat   = (const float*)d_in[0];
    const float* coords = (const float*)d_in[1];
    const float* params = (const float*)d_in[2];
    const float* gauss  = (const float*)d_in[3];
    float* out = (float*)d_out;

    // 192 KB dynamic LDS (<= 320 KB per WGP on CDNA5); set limit every call
    // (deterministic, not a stream op, safe under graph capture).
    hipFuncSetAttribute((const void*)ImplicitPointHead_kernel,
                        hipFuncAttributeMaxDynamicSharedMemorySize, LDS_BYTES);

    dim3 grid(2048 / PTILE, 128);   // (point tiles, instances)
    ImplicitPointHead_kernel<<<grid, 256, LDS_BYTES, stream>>>(
        feat, coords, params, gauss, out);
}
